// TaskAlignedAssigner_56736517980201
// MI455X (gfx1250) — matrix operand order repaired and checked by reading, not compiled, not run on hardware
//
#include <hip/hip_runtime.h>
#include <hip/hip_fp16.h>
#include <stdint.h>
#include <limits.h>

typedef __attribute__((ext_vector_type(16))) _Float16 v16h;
typedef __attribute__((ext_vector_type(8)))  float    v8f;

#define BS      32
#define NP      8400
#define NC      80
#define NCP     96                            // NC padded to 3 K-tiles of 32
#define NG      128
#define TOPK_K  64
#define GTILE   16
#define NWAVES  8
#define BLOCK   256
#define PCHUNK  (NWAVES * 16)                 // 128 points per chunk
#define NCHUNK  ((NP + PCHUNK - 1) / PCHUNK)  // 66
#define CANDCAP 128
#define EPSF    1e-7f

// IoU + align metric (ALPHA=0.5 -> sqrt, BETA=2.0 -> square).
// v_rcp_f32 / raw v_sqrt_f32 instead of IEEE macro-sequences; identical
// instructions in every pass, so selection stays bit-consistent.
__device__ __forceinline__ float align_metric(float cls, float4 pb, float4 gb) {
  float w1 = pb.z - pb.x;
  float h1 = pb.w - pb.y + EPSF;
  float w2 = gb.z - gb.x;
  float h2 = gb.w - gb.y + EPSF;
  float iw = fmaxf(fminf(pb.z, gb.z) - fmaxf(pb.x, gb.x), 0.0f);
  float ih = fmaxf(fminf(pb.w, gb.w) - fmaxf(pb.y, gb.y), 0.0f);
  float inter = iw * ih;
  float uni = w1 * h1 + w2 * h2 - inter + EPSF;
  float iou = inter * __builtin_amdgcn_rcpf(uni);
  float t = fabsf(iou) + 1e-5f;
  return __builtin_amdgcn_sqrtf(cls + 1e-5f) * (t * t);
}

__global__ void taa_init(int* ws, int n) {
  int i = blockIdx.x * blockDim.x + threadIdx.x;
  if (i < n) ws[i] = -1;
}

// One-time sigmoid precompute: f32 scores -> f16, padded NC=80 -> 96 classes.
// Removes all transcendentals/divisions from the multi-pass selection loop.
__global__ void taa_sigmoid(const float* __restrict__ scores,
                            _Float16* __restrict__ sig16) {
  int i = blockIdx.x * blockDim.x + threadIdx.x;  // over BS*NP points
  if (i >= BS * NP) return;
  const float* src = scores + (size_t)i * NC;
  _Float16* dst = sig16 + (size_t)i * NCP;
  #pragma unroll 4
  for (int c = 0; c < NC; ++c) {
    float s = __builtin_amdgcn_rcpf(1.0f + __expf(-src[c]));
    dst[c] = (_Float16)s;
  }
  #pragma unroll
  for (int c = NC; c < NCP; ++c) dst[c] = (_Float16)0.0f;
}

// One block per (batch, tile of 16 GTs). WMMA computes cls = OneHot(labels) @ sig
// (the per-GT class gather as a one-hot matmul) for 16 GTs x 16 points per wave
// per chunk. 4 radix passes find the exact bit pattern of the 64th-largest align
// value per GT row; the marking pass records strictly-greater winners via
// atomicMax and collects threshold-equal candidates for index-ordered tie-break.
__global__ __launch_bounds__(BLOCK) void taa_select(
    const _Float16* __restrict__ sig16,         // [BS][NP][NCP]
    const float* __restrict__ pd_bboxes,        // [BS][NP][4]
    const int*   __restrict__ gt_labels,        // [BS][NG]
    const float* __restrict__ gt_bboxes,        // [BS][NG][4]
    const unsigned char* __restrict__ mask_gt,  // [BS][NG]
    int* __restrict__ ws_idx)                   // [BS][NP], pre-initialized to -1
{
  const int b    = blockIdx.y;
  const int g0   = blockIdx.x * GTILE;
  const int tid  = threadIdx.x;
  const int lane = tid & 31;
  const int wave = tid >> 5;
  const int ncol  = lane & 15;   // column (point) index within 16-wide tile
  const int khalf = lane >> 4;   // lane half selects K/M sub-block per WMMA layout

  __shared__ float4   s_gtbox[GTILE];
  __shared__ int      s_lbl[GTILE];
  __shared__ int      s_valid[GTILE];
  __shared__ unsigned s_hist[GTILE][256];
  __shared__ unsigned s_prefix[GTILE];
  __shared__ unsigned s_k[GTILE];
  __shared__ unsigned s_T[GTILE];
  __shared__ int      s_kfin[GTILE];
  __shared__ int      s_cand[GTILE][CANDCAP];
  __shared__ int      s_ccnt[GTILE];

  // Stage the 16 GT boxes into LDS via CDNA5 async-to-LDS DMA (ASYNCcnt-tracked).
  if (tid < GTILE) {
    const float4* src = (const float4*)gt_bboxes + (size_t)b * NG + (g0 + tid);
    unsigned lds_off = (unsigned)(uintptr_t)(&s_gtbox[tid]);
    unsigned long long gaddr = (unsigned long long)(uintptr_t)src;
    asm volatile("global_load_async_to_lds_b128 %0, %1, off"
                 :: "v"(lds_off), "v"(gaddr) : "memory");
    s_lbl[tid]    = gt_labels[b * NG + g0 + tid];
    s_valid[tid]  = (mask_gt[b * NG + g0 + tid] != 0) ? 1 : 0;
    s_prefix[tid] = 0u;
    s_k[tid]      = (unsigned)TOPK_K;
    s_ccnt[tid]   = 0;
  }
  asm volatile("s_wait_asynccnt 0x0" ::: "memory");
  __syncthreads();

  // Per-thread register copies of loop-invariant row state (the compiler cannot
  // hoist these LDS reads itself because of the LDS atomics in the same loops).
  int vrow[8];
  #pragma unroll
  for (int r = 0; r < 8; ++r) vrow[r] = s_valid[r + 8 * khalf];

  // One-hot A fragments (16 GTs x 96 classes, 3 K-tiles of 32), kept in VGPRs.
  // A 16x32 in this code's contiguous k order (B matches): lane -> M = lane%16; element e -> K = e + 16*khalf.
  const int my_lbl = s_lbl[ncol];
  v16h afrag[3];
  #pragma unroll
  for (int kt = 0; kt < 3; ++kt) {
    v16h a;
    #pragma unroll
    for (int e = 0; e < 16; ++e) {
      int K = kt * 32 + e + 16 * khalf;
      a[e] = (_Float16)((K == my_lbl) ? 1.0f : 0.0f);
    }
    afrag[kt] = a;
  }

  // Recompute align fragment for one chunk: 8 rows (GTs) x this lane's point.
  // All three B fragments (32B v16h each) are issued up front so the compiler
  // can clause the loads and overlap latency with the WMMA chain.
  auto compute_chunk = [&](int chunk, float av[8], int& pcol) {
    pcol = chunk * PCHUNK + wave * 16 + ncol;
    int psafe = (pcol < NP) ? pcol : (NP - 1);
    const _Float16* srow = sig16 + ((size_t)b * NP + psafe) * NCP;
    __builtin_prefetch(srow + (size_t)PCHUNK * NCP, 0, 3);  // global_prefetch_b8
    // B 32x16, same contiguous k order as A: lane -> N = lane%16; element e -> K = e + 16*khalf.
    v16h b0 = *(const v16h*)(srow + 0  + 16 * khalf);
    v16h b1 = *(const v16h*)(srow + 32 + 16 * khalf);
    v16h b2 = *(const v16h*)(srow + 64 + 16 * khalf);
    float4 pb = ((const float4*)pd_bboxes)[(size_t)b * NP + psafe];
    v8f c = {};
    c = __builtin_amdgcn_wmma_f32_16x16x32_f16(false, afrag[0], false, b0,
                                               (short)0, c, false, false);
    c = __builtin_amdgcn_wmma_f32_16x16x32_f16(false, afrag[1], false, b1,
                                               (short)0, c, false, false);
    c = __builtin_amdgcn_wmma_f32_16x16x32_f16(false, afrag[2], false, b2,
                                               (short)0, c, false, false);
    #pragma unroll
    for (int r = 0; r < 8; ++r) {
      int row = r + 8 * khalf;  // C layout: VGPR r -> M = r (+8 for upper lanes)
      av[r] = align_metric(c[r], pb, s_gtbox[row]);
    }
  };

  // 4 radix passes (8 bits each, MSB first): exact bit pattern of the 64th-largest
  // align value per row. All align values are positive, so float bits order as uint.
  #pragma unroll 1
  for (int pass = 0; pass < 4; ++pass) {
    const int shift = 24 - 8 * pass;
    const unsigned msk = (pass == 0) ? 0u : (0xFFFFFFFFu << (shift + 8));
    for (int i = tid; i < GTILE * 256; i += BLOCK) ((unsigned*)s_hist)[i] = 0u;
    __syncthreads();
    // Hoist this pass's per-row prefixes into registers (invariant until the
    // next barrier; LDS atomics below would otherwise block the hoist).
    unsigned pfx[8];
    #pragma unroll
    for (int r = 0; r < 8; ++r) pfx[r] = s_prefix[r + 8 * khalf];
    #pragma unroll 1
    for (int chunk = 0; chunk < NCHUNK; ++chunk) {
      float av[8]; int pcol;
      compute_chunk(chunk, av, pcol);
      if (pcol < NP) {
        #pragma unroll
        for (int r = 0; r < 8; ++r) {
          int row = r + 8 * khalf;
          unsigned v = __float_as_uint(av[r]);
          if ((v & msk) == pfx[r])
            atomicAdd(&s_hist[row][(v >> shift) & 255], 1u);
        }
      }
    }
    __syncthreads();
    if (tid < GTILE && s_valid[tid]) {
      unsigned k = s_k[tid];
      int sel = 0;
      for (int bin = 255; bin >= 0; --bin) {
        unsigned cnt = s_hist[tid][bin];
        if (cnt >= k) { sel = bin; break; }
        k -= cnt;
      }
      s_prefix[tid] |= ((unsigned)sel) << shift;
      s_k[tid] = k;
    }
    __syncthreads();
  }
  if (tid < GTILE) {
    s_T[tid]    = s_prefix[tid];
    s_kfin[tid] = s_valid[tid] ? (int)s_k[tid] : 0;
  }
  __syncthreads();

  // Register copies of per-row thresholds for the marking pass.
  unsigned Tr[8];
  #pragma unroll
  for (int r = 0; r < 8; ++r) Tr[r] = s_T[r + 8 * khalf];

  // Marking pass: strictly-greater values are definite top-k members; values equal
  // to the threshold bit pattern become tie-break candidates (with point index).
  int* wsb = ws_idx + b * NP;
  #pragma unroll 1
  for (int chunk = 0; chunk < NCHUNK; ++chunk) {
    float av[8]; int pcol;
    compute_chunk(chunk, av, pcol);
    if (pcol < NP) {
      #pragma unroll
      for (int r = 0; r < 8; ++r) {
        int row = r + 8 * khalf;
        if (vrow[r]) {
          unsigned v = __float_as_uint(av[r]);
          if (v > Tr[r]) {
            atomicMax(&wsb[pcol], g0 + row);
          } else if (v == Tr[r]) {
            int pos = atomicAdd(&s_ccnt[row], 1);
            if (pos < CANDCAP) s_cand[row][pos] = pcol;
          }
        }
      }
    }
  }
  __syncthreads();

  // Tie-break: select the k_fin smallest point indices among threshold-equal
  // candidates (jax top_k breaks ties by lowest index first).
  if (tid < GTILE && s_kfin[tid] > 0) {
    const int row = tid;
    int n = s_ccnt[row];
    if (n > CANDCAP) n = CANDCAP;
    int need = s_kfin[row];
    int last = -1;
    for (int s = 0; s < need; ++s) {
      int best = INT_MAX;
      for (int j = 0; j < n; ++j) {
        int idx = s_cand[row][j];
        if (idx > last && idx < best) best = idx;
      }
      if (best == INT_MAX) break;
      atomicMax(&wsb[best], g0 + row);
      last = best;
    }
  }
}

__global__ void taa_finalize(const int* __restrict__ ws,
                             const int* __restrict__ gt_labels,
                             const float* __restrict__ gt_bboxes,
                             float* __restrict__ out)
{
  int i = blockIdx.x * blockDim.x + threadIdx.x;
  if (i >= BS * NP) return;
  int b = i / NP;
  int m = ws[i];
  int assigned = (m >= 0) ? 1 : 0;
  int idx = assigned ? m : 0;
  int lbl = assigned ? gt_labels[b * NG + idx] : 0;

  float* out_lab = out;                              // [BS*NP]
  float* out_box = out + (size_t)BS * NP;            // [BS*NP*4]
  float* out_sco = out + (size_t)BS * NP * 5;        // [BS*NP*80]
  float* out_fg  = out + (size_t)BS * NP * 85;       // [BS*NP]
  float* out_idx = out + (size_t)BS * NP * 86;       // [BS*NP]

  out_lab[i] = (float)lbl;
  float4 gb = make_float4(0.f, 0.f, 0.f, 0.f);
  if (assigned) gb = ((const float4*)gt_bboxes)[(size_t)b * NG + idx];
  ((float4*)out_box)[i] = gb;
  float* srow = out_sco + (size_t)i * NC;
  for (int c = 0; c < NC; ++c)
    srow[c] = (assigned && c == lbl) ? 1.0f : 0.0f;
  out_fg[i]  = assigned ? 1.0f : 0.0f;
  out_idx[i] = (float)idx;
}

extern "C" void kernel_launch(void* const* d_in, const int* in_sizes, int n_in,
                              void* d_out, int out_size, void* d_ws, size_t ws_size,
                              hipStream_t stream) {
  const float* pd_scores = (const float*)d_in[0];
  const float* pd_bboxes = (const float*)d_in[1];
  // d_in[2] = anchor_points (unused by reference assignment)
  const int*   gt_labels = (const int*)d_in[3];
  const float* gt_bboxes = (const float*)d_in[4];
  const unsigned char* mask_gt = (const unsigned char*)d_in[5];

  // Workspace layout: [BS*NP] int best-GT index, then [BS*NP*96] f16 sigmoid
  // scores (~52.7 MB total; L2-resident on MI455X's 192 MB L2).
  int* ws_idx = (int*)d_ws;
  _Float16* sig16 = (_Float16*)((char*)d_ws + (size_t)BS * NP * sizeof(int));

  int n = BS * NP;
  taa_init<<<(n + 255) / 256, 256, 0, stream>>>(ws_idx, n);
  taa_sigmoid<<<(n + 255) / 256, 256, 0, stream>>>(pd_scores, sig16);
  dim3 grid(NG / GTILE, BS);
  taa_select<<<grid, BLOCK, 0, stream>>>(sig16, pd_bboxes, gt_labels,
                                         gt_bboxes, mask_gt, ws_idx);
  taa_finalize<<<(n + 255) / 256, 256, 0, stream>>>(ws_idx, gt_labels, gt_bboxes,
                                                    (float*)d_out);
}
